// AttnBlock_34608846471454
// MI455X (gfx1250) — matrix operand order repaired and checked by reading, not compiled, not run on hardware
//
#include <hip/hip_runtime.h>
#include <hip/hip_bf16.h>

// ---------- types ----------
typedef __attribute__((ext_vector_type(16))) __bf16 v16bf;
typedef __attribute__((ext_vector_type(8)))  float  v8f;
typedef __attribute__((ext_vector_type(4)))  unsigned int v4u;
typedef __attribute__((ext_vector_type(8)))  int v8i;
typedef __attribute__((ext_vector_type(4)))  int v4i;

union Frag { v16bf v; uint4 q[2]; };

__device__ __forceinline__ unsigned short f2bf(float f) {
    unsigned int u = __builtin_bit_cast(unsigned int, f);
    unsigned int r = u + 0x7FFFu + ((u >> 16) & 1u);
    return (unsigned short)(r >> 16);
}
__device__ __forceinline__ float bf2f(unsigned short h) {
    return __builtin_bit_cast(float, (unsigned int)h << 16);
}

// ---------- problem constants ----------
#define BATCH 4096
#define SEQ   10
#define FEAT  512
#define DMOD  512
#define HEADS 4
#define MROWS (BATCH * SEQ)          // 40960
#define NQKV  (3 * HEADS * DMOD)     // 6144
#define HD    (HEADS * DMOD)         // 2048

// ---------- workspace layout (bytes) ----------
static constexpr size_t SZ_XB    = (size_t)MROWS * FEAT * 2;
static constexpr size_t SZ_XMEAN = (size_t)BATCH * FEAT * 4;
static constexpr size_t SZ_GATE  = (size_t)BATCH * HEADS * 4;
static constexpr size_t SZ_WQKVT = (size_t)3 * HEADS * FEAT * DMOD * 2;
static constexpr size_t SZ_BQKV  = (size_t)NQKV * 4;
static constexpr size_t SZ_WT    = (size_t)HD * DMOD * 2;     // 2048x512 bf16
static constexpr size_t SZ_QKV   = (size_t)3 * BATCH * HEADS * SEQ * DMOD * 2;
static constexpr size_t SZ_HCAT  = (size_t)MROWS * HD * 2;
static constexpr size_t SZ_ROWF  = (size_t)MROWS * DMOD * 4;

static constexpr size_t OFF_XB    = 0;
static constexpr size_t OFF_XMEAN = OFF_XB    + SZ_XB;
static constexpr size_t OFF_GATE  = OFF_XMEAN + SZ_XMEAN;
static constexpr size_t OFF_WQKVT = OFF_GATE  + SZ_GATE;
static constexpr size_t OFF_BQKV  = OFF_WQKVT + SZ_WQKVT;
static constexpr size_t OFF_WPT   = OFF_BQKV  + SZ_BQKV;
static constexpr size_t OFF_W1T   = OFF_WPT   + SZ_WT;
static constexpr size_t OFF_W2T   = OFF_W1T   + SZ_WT;
static constexpr size_t OFF_QKV   = OFF_W2T   + SZ_WT;
static constexpr size_t OFF_F1    = OFF_QKV;               // alias: qkv dead before ffn1
static constexpr size_t OFF_HCAT  = OFF_QKV   + SZ_QKV;
static constexpr size_t OFF_PROJ  = OFF_HCAT  + SZ_HCAT;
static constexpr size_t OFF_FFN   = OFF_PROJ;              // alias: proj dead before ffn2
static constexpr size_t OFF_X1    = OFF_PROJ  + SZ_ROWF;
static constexpr size_t OFF_X1B   = OFF_X1    + SZ_ROWF;

// ---------- small prep kernels ----------
__global__ void k_cvt_bf16(const float* __restrict__ in, unsigned short* __restrict__ out, int n4) {
    int i = blockIdx.x * 256 + threadIdx.x;
    if (i < n4) {
        float4 v = ((const float4*)in)[i];
        ushort4 o;
        o.x = f2bf(v.x); o.y = f2bf(v.y); o.z = f2bf(v.z); o.w = f2bf(v.w);
        ((ushort4*)out)[i] = o;
    }
}

__global__ void k_xmean(const float* __restrict__ x, float* __restrict__ xm) {
    int idx = blockIdx.x * 256 + threadIdx.x;       // over B*F
    int b = idx >> 9, f = idx & 511;
    const float* p = x + (size_t)b * SEQ * FEAT + f;
    float s = 0.f;
#pragma unroll
    for (int i = 0; i < SEQ; ++i) s += p[i * FEAT];
    xm[idx] = s * (1.0f / SEQ);
}

// WqkvT rows n = (qs*H+h)*D + d, cols f ;  n*K+f
__global__ void k_prep_qkvT(const float* __restrict__ Wq, const float* __restrict__ Wk,
                            const float* __restrict__ Wv, unsigned short* __restrict__ wt) {
    int idx = blockIdx.x * 256 + threadIdx.x;       // over H*F*D = 4*512*512
    int h = idx >> 18;
    int rem = idx & 262143;
    int f = rem >> 9, d = rem & 511;
    size_t dst = ((size_t)h * 512 + d) * 512 + f;
    wt[dst]              = f2bf(Wq[idx]);
    wt[dst + 1048576]    = f2bf(Wk[idx]);
    wt[dst + 2097152]    = f2bf(Wv[idx]);
}

__global__ void k_prep_qkv_bias(const float* __restrict__ bq, const float* __restrict__ bk,
                                const float* __restrict__ bv, float* __restrict__ bias) {
    int n = blockIdx.x * 256 + threadIdx.x;
    if (n < NQKV) {
        int qs = n >> 11, h = (n >> 9) & 3, d = n & 511;
        const float* src = (qs == 0) ? bq : (qs == 1) ? bk : bv;
        bias[n] = src[h * 512 + d];
    }
}

// out[n*K + k] = bf16(in[k*N + n])
__global__ void k_transposeT(const float* __restrict__ in, unsigned short* __restrict__ out,
                             int K, int N) {
    int idx = blockIdx.x * 256 + threadIdx.x;
    if (idx < K * N) {
        int k = idx / N, n = idx - k * N;
        out[(size_t)n * K + k] = f2bf(in[idx]);
    }
}

__global__ __launch_bounds__(128) void k_gate(const float* __restrict__ xmean,
                                              const float* __restrict__ Wg,
                                              const float* __restrict__ bg,
                                              float* __restrict__ gate) {
    const int b = blockIdx.x;
    const int t = threadIdx.x;
    const int h = t >> 5, lane = t & 31;
    __shared__ float sh[HEADS];
    float acc = 0.f;
    for (int f = lane; f < FEAT; f += 32) acc += xmean[(size_t)b * FEAT + f] * Wg[f * HEADS + h];
#pragma unroll
    for (int off = 16; off > 0; off >>= 1) acc += __shfl_down(acc, off, 32);
    if (lane == 0) sh[h] = acc + bg[h];
    __syncthreads();
    if (t == 0) {
        float mx = fmaxf(fmaxf(sh[0], sh[1]), fmaxf(sh[2], sh[3]));
        float e0 = __expf(sh[0] - mx), e1 = __expf(sh[1] - mx);
        float e2 = __expf(sh[2] - mx), e3 = __expf(sh[3] - mx);
        float inv = 1.f / (e0 + e1 + e2 + e3);
        gate[b * 4 + 0] = e0 * inv; gate[b * 4 + 1] = e1 * inv;
        gate[b * 4 + 2] = e2 * inv; gate[b * 4 + 3] = e3 * inv;
    }
}

// ---------- TDM: 2D tile DMA global -> LDS (CDNA5 tensor_load_to_lds) ----------
// Loads tile_rows rows of 64 bytes each (row stride = row_stride_bytes) into LDS at
// lds_off, inserting 16B of LDS padding after every 64B row (pad_interval=16 DWORDs,
// pad_amount=4 DWORDs) -> padded LDS row stride of 80B, matching the compute layout.
// Descriptor fields per CDNA5 ISA ch.8 (D# group0/group1); data_size = 4 bytes.
// This toolchain's builtin takes 6 args: (g0 v4u, g1 v8i, g2 v4i, g3 v4i, g4 v8i, cpol).
__device__ __forceinline__ void tdm_load_tile64(unsigned lds_off, const void* gptr,
                                                unsigned tile_rows,
                                                unsigned long long row_stride_bytes) {
    unsigned long long ga = (unsigned long long)gptr;
    unsigned long long s0 = row_stride_bytes >> 2;     // dim0 stride in dwords
    const unsigned dim0_dw = 16;                       // 64B tile row = 16 dwords

    v4u g0;
    g0[0] = 1u;                                        // count=1, user descriptor
    g0[1] = lds_off;                                   // LDS byte address
    g0[2] = (unsigned)(ga & 0xffffffffu);              // global_addr[31:0]
    g0[3] = (unsigned)((ga >> 32) & 0x1ffffffu) | (2u << 30);  // addr[56:32] | type=2

    v8i g1;
    // data_size=4B(2), pad_enable, pad_interval=16dw(3), pad_amount=4dw(3)
    g1[0] = (int)((2u << 16) | (1u << 20) | (3u << 22) | (3u << 25));
    unsigned td0 = (unsigned)s0;                       // tensor_dim0 (row fully in bounds)
    unsigned td1 = tile_rows;                          // tensor_dim1
    g1[1] = (int)((td0 & 0xffffu) << 16);              // abar_addr=0 | tensor_dim0[15:0]
    g1[2] = (int)(((td0 >> 16) & 0xffffu) | ((td1 & 0xffffu) << 16));
    g1[3] = (int)(((td1 >> 16) & 0xffffu) | (dim0_dw << 16));     // tile_dim0=16
    g1[4] = (int)(tile_rows & 0xffffu);                // tile_dim1 ; tile_dim2=0
    g1[5] = (int)(s0 & 0xffffffffu);                   // tensor_dim0_stride[31:0]
    g1[6] = (int)((s0 >> 32) & 0xffffu);               // stride[47:32] | d1stride lo=0
    g1[7] = 0;

    v4i z4 = {0, 0, 0, 0};                             // 2D tensor: groups 2/3 unused
    v8i z8 = {0, 0, 0, 0, 0, 0, 0, 0};
    __builtin_amdgcn_tensor_load_to_lds(g0, g1, z4, z4, z8, 0);
}

// ---------- WMMA GEMM:  C[M,N] = A[M,K](bf16) * Bt[N,K](bf16)^T + bias ----------
// TDM double-buffered tiles; 8 waves x 32x64 wave-tile; 8 v_wmma per K-step per wave.
// MODE 0: bf16 out, scattered to qkv[3][B][H][S][D]
// MODE 1: f32 out, row-major ldc=N
// MODE 2: relu -> bf16 out, row-major ldc=N
#define BM 128
#define BN 128
#define BK 32
#define LDS_S 40   // padded LDS row stride (elems) = 64B data + 16B TDM pad

template <int MODE>
__global__ __launch_bounds__(256) void k_gemm(const unsigned short* __restrict__ A,
                                              const unsigned short* __restrict__ Bt,
                                              const float* __restrict__ bias,
                                              void* __restrict__ out,
                                              int M, int N, int K) {
    __shared__ alignas(16) unsigned short As[2][BM * LDS_S];
    __shared__ alignas(16) unsigned short Bs[2][BN * LDS_S];
    const int tid  = threadIdx.x;
    const int lane = tid & 31;
    const int wid  = tid >> 5;
    const int wm   = wid >> 1;          // 0..3  (32 rows each)
    const int wn   = wid & 1;           // 0..1  (64 cols each)
    const int m0   = blockIdx.x * BM;
    const int n0   = blockIdx.y * BN;
    const int r    = lane & 15;
    const int kh   = lane >> 4;         // lane half selects K sub-chunks

    const unsigned aoff0 = (unsigned)(unsigned long long)(void*)&As[0][0];
    const unsigned aoff1 = (unsigned)(unsigned long long)(void*)&As[1][0];
    const unsigned boff0 = (unsigned)(unsigned long long)(void*)&Bs[0][0];
    const unsigned boff1 = (unsigned)(unsigned long long)(void*)&Bs[1][0];
    const unsigned long long strideA = (unsigned long long)K * 2ull;

    v8f acc[2][4];
#pragma unroll
    for (int i = 0; i < 2; ++i)
#pragma unroll
        for (int j = 0; j < 4; ++j)
            acc[i][j] = v8f{0.f, 0.f, 0.f, 0.f, 0.f, 0.f, 0.f, 0.f};

    const int ksteps = K / BK;

    // prologue: TDM-load K-tile 0 into buffer 0 (one wave issues; TDM ignores EXEC)
    if (wid == 0) {
        tdm_load_tile64(aoff0, A  + (size_t)m0 * K, BM, strideA);
        tdm_load_tile64(boff0, Bt + (size_t)n0 * K, BN, strideA);
        __builtin_amdgcn_s_wait_tensorcnt(0);
    }
    __syncthreads();

    for (int kt = 0; kt < ksteps; ++kt) {
        const int cur = kt & 1;
        // kick off DMA of next K-tile into the other buffer while computing this one
        if (wid == 0 && (kt + 1) < ksteps) {
            const int k1 = (kt + 1) * BK;
            tdm_load_tile64(cur ? aoff0 : aoff1, A  + (size_t)m0 * K + k1, BM, strideA);
            tdm_load_tile64(cur ? boff0 : boff1, Bt + (size_t)n0 * K + k1, BN, strideA);
        }
        const unsigned short* Ab = As[cur];
        const unsigned short* Bb = Bs[cur];

        // A fragments: lanes 0-15 hold K 0-7 & 16-23, lanes 16-31 hold K 8-15 & 24-31
        Frag af[2], bf[4];
#pragma unroll
        for (int mt = 0; mt < 2; ++mt) {
            int e = (wm * 32 + mt * 16 + r) * LDS_S + kh * 8;
            af[mt].q[0] = *(const uint4*)(Ab + e);
            af[mt].q[1] = *(const uint4*)(Ab + e + 16);
        }
        // B fragments: lanes 0-15 hold K 0-7 & 16-23 (col n), lanes 16-31 hold K 8-15 & 24-31
#pragma unroll
        for (int nt = 0; nt < 4; ++nt) {
            int e = (wn * 64 + nt * 16 + r) * LDS_S + kh * 8;
            bf[nt].q[0] = *(const uint4*)(Bb + e);
            bf[nt].q[1] = *(const uint4*)(Bb + e + 16);
        }
#pragma unroll
        for (int mt = 0; mt < 2; ++mt)
#pragma unroll
            for (int nt = 0; nt < 4; ++nt)
                acc[mt][nt] = __builtin_amdgcn_wmma_f32_16x16x32_bf16(
                    false, af[mt].v, false, bf[nt].v, (short)0, acc[mt][nt], false, false);

        // make next buffer visible to everyone before the buffer flip
        if (wid == 0) __builtin_amdgcn_s_wait_tensorcnt(0);
        __syncthreads();
    }

    // epilogue: C/D layout -> lanes 0-15: M=i, N=lane ; lanes 16-31: M=8+i, N=lane-16
#pragma unroll
    for (int mt = 0; mt < 2; ++mt) {
#pragma unroll
        for (int nt = 0; nt < 4; ++nt) {
            int gn = n0 + wn * 64 + nt * 16 + r;
            float bv = bias[gn];
#pragma unroll
            for (int i = 0; i < 8; ++i) {
                int gm = m0 + wm * 32 + mt * 16 + kh * 8 + i;
                float v = acc[mt][nt][i] + bv;
                if (MODE == 0) {
                    int qs = gn >> 11, h = (gn >> 9) & 3, d = gn & 511;
                    int bb = gm / SEQ, s = gm - bb * SEQ;
                    ((unsigned short*)out)[((((size_t)qs * BATCH + bb) * HEADS + h) * SEQ + s) * DMOD + d] = f2bf(v);
                } else if (MODE == 1) {
                    ((float*)out)[(size_t)gm * N + gn] = v;
                } else {
                    ((unsigned short*)out)[(size_t)gm * N + gn] = f2bf(v > 0.f ? v : 0.f);
                }
            }
        }
    }
}

// ---------- attention (S=10, per (b,h) block; <1% of FLOPs -> VALU) ----------
__global__ __launch_bounds__(128) void k_attention(const unsigned short* __restrict__ qkv,
                                                   const float* __restrict__ gate,
                                                   unsigned short* __restrict__ hcat) {
    __shared__ alignas(16) unsigned short Qs[SEQ * DMOD];
    __shared__ alignas(16) unsigned short Ks[SEQ * DMOD];
    __shared__ alignas(16) unsigned short Vs[SEQ * DMOD];
    __shared__ float P[SEQ][SEQ + 2];
    const int bh = blockIdx.x;
    const int b = bh >> 2, h = bh & 3;
    const int tid = threadIdx.x;
    const size_t stride_qs = (size_t)BATCH * HEADS * SEQ * DMOD;
    const size_t base = ((size_t)b * HEADS + h) * SEQ * DMOD;

    for (int i = tid; i < (SEQ * DMOD) / 8; i += 128) {
        ((uint4*)Qs)[i] = ((const uint4*)(qkv + base))[i];
        ((uint4*)Ks)[i] = ((const uint4*)(qkv + base + stride_qs))[i];
        ((uint4*)Vs)[i] = ((const uint4*)(qkv + base + 2 * stride_qs))[i];
    }
    __syncthreads();

    if (tid < SEQ * SEQ) {
        int i = tid / SEQ, j = tid - SEQ * (tid / SEQ);
        float acc = 0.f;
        for (int d = 0; d < DMOD; ++d) acc += bf2f(Qs[i * DMOD + d]) * bf2f(Ks[j * DMOD + d]);
        P[i][j] = acc * 0.04419417382415922f;   // 1/sqrt(512)
    }
    __syncthreads();

    if (tid < SEQ) {
        float mx = -1e30f;
        for (int j = 0; j < SEQ; ++j) mx = fmaxf(mx, P[tid][j]);
        float e[SEQ], sm = 0.f;
        for (int j = 0; j < SEQ; ++j) { e[j] = __expf(P[tid][j] - mx); sm += e[j]; }
        float inv = 1.f / sm;
        for (int j = 0; j < SEQ; ++j) P[tid][j] = e[j] * inv;
    }
    __syncthreads();

    const float g = gate[b * HEADS + h];
    for (int idx = tid; idx < SEQ * DMOD; idx += 128) {
        int s = idx >> 9, d = idx & 511;
        float acc = 0.f;
#pragma unroll
        for (int t = 0; t < SEQ; ++t) acc += P[s][t] * bf2f(Vs[t * DMOD + d]);
        hcat[((size_t)(b * SEQ + s)) * HD + h * DMOD + d] = f2bf(acc * g);
    }
}

// ---------- residual + layernorm (row of 512) ----------
template <bool WRITE_BF16>
__global__ __launch_bounds__(128) void k_ln(const float* __restrict__ pa, const float* __restrict__ pb,
                                            const float* __restrict__ g, const float* __restrict__ be,
                                            float* __restrict__ of, unsigned short* __restrict__ ob) {
    __shared__ float red[128], red2[128];
    const int row = blockIdx.x;
    const int t = threadIdx.x;
    const float* ap = pa + (size_t)row * DMOD;
    const float* bp2 = pb + (size_t)row * DMOD;
    float v[4], s = 0.f, s2 = 0.f;
#pragma unroll
    for (int i = 0; i < 4; ++i) {
        int c = t + i * 128;
        v[i] = ap[c] + bp2[c];
        s += v[i]; s2 += v[i] * v[i];
    }
    red[t] = s; red2[t] = s2;
    __syncthreads();
    for (int off = 64; off > 0; off >>= 1) {
        if (t < off) { red[t] += red[t + off]; red2[t] += red2[t + off]; }
        __syncthreads();
    }
    float mean = red[0] * (1.f / DMOD);
    float var  = red2[0] * (1.f / DMOD) - mean * mean;
    float inv  = rsqrtf(var + 1e-5f);
#pragma unroll
    for (int i = 0; i < 4; ++i) {
        int c = t + i * 128;
        float y = (v[i] - mean) * inv * g[c] + be[c];
        of[(size_t)row * DMOD + c] = y;
        if (WRITE_BF16) ob[(size_t)row * DMOD + c] = f2bf(y);
    }
}

// ---------- launcher ----------
extern "C" void kernel_launch(void* const* d_in, const int* in_sizes, int n_in,
                              void* d_out, int out_size, void* d_ws, size_t ws_size,
                              hipStream_t stream) {
    const float* x   = (const float*)d_in[0];
    const float* Wq  = (const float*)d_in[1];
    const float* bq  = (const float*)d_in[2];
    const float* Wk  = (const float*)d_in[3];
    const float* bk  = (const float*)d_in[4];
    const float* Wv  = (const float*)d_in[5];
    const float* bv  = (const float*)d_in[6];
    const float* Wg  = (const float*)d_in[7];
    const float* bg  = (const float*)d_in[8];
    const float* Wp  = (const float*)d_in[9];
    const float* bp  = (const float*)d_in[10];
    const float* W1  = (const float*)d_in[11];
    const float* b1  = (const float*)d_in[12];
    const float* W2  = (const float*)d_in[13];
    const float* b2  = (const float*)d_in[14];
    const float* g1  = (const float*)d_in[15];
    const float* be1 = (const float*)d_in[16];
    const float* g2  = (const float*)d_in[17];
    const float* be2 = (const float*)d_in[18];
    float* out = (float*)d_out;

    char* ws = (char*)d_ws;
    unsigned short* xb    = (unsigned short*)(ws + OFF_XB);
    float*          xmean = (float*)(ws + OFF_XMEAN);
    float*          gate  = (float*)(ws + OFF_GATE);
    unsigned short* wqkvT = (unsigned short*)(ws + OFF_WQKVT);
    float*          bqkv  = (float*)(ws + OFF_BQKV);
    unsigned short* wpT   = (unsigned short*)(ws + OFF_WPT);
    unsigned short* w1T   = (unsigned short*)(ws + OFF_W1T);
    unsigned short* w2T   = (unsigned short*)(ws + OFF_W2T);
    unsigned short* qkv   = (unsigned short*)(ws + OFF_QKV);
    unsigned short* f1    = (unsigned short*)(ws + OFF_F1);
    unsigned short* hcat  = (unsigned short*)(ws + OFF_HCAT);
    float*          proj  = (float*)(ws + OFF_PROJ);
    float*          ffn   = (float*)(ws + OFF_FFN);
    float*          x1    = (float*)(ws + OFF_X1);
    unsigned short* x1b   = (unsigned short*)(ws + OFF_X1B);

    // prep: bf16 activations + transposed bf16 weights + gate inputs
    k_cvt_bf16<<<(MROWS * FEAT / 4 + 255) / 256, 256, 0, stream>>>(x, xb, MROWS * FEAT / 4);
    k_xmean<<<(BATCH * FEAT) / 256, 256, 0, stream>>>(x, xmean);
    k_prep_qkvT<<<(HEADS * FEAT * DMOD) / 256, 256, 0, stream>>>(Wq, Wk, Wv, wqkvT);
    k_prep_qkv_bias<<<(NQKV + 255) / 256, 256, 0, stream>>>(bq, bk, bv, bqkv);
    k_transposeT<<<(HD * DMOD) / 256, 256, 0, stream>>>(Wp, wpT, HD, DMOD);   // [2048,512] -> [512,2048]
    k_transposeT<<<(DMOD * HD) / 256, 256, 0, stream>>>(W1, w1T, DMOD, HD);   // [512,2048] -> [2048,512]
    k_transposeT<<<(HD * DMOD) / 256, 256, 0, stream>>>(W2, w2T, HD, DMOD);   // [2048,512] -> [512,2048]
    k_gate<<<BATCH, 128, 0, stream>>>(xmean, Wg, bg, gate);

    // QKV: [40960,512] x [512,6144]  -> qkv bf16 [3,B,H,S,D]
    k_gemm<0><<<dim3(MROWS / BM, NQKV / BN), 256, 0, stream>>>(xb, wqkvT, bqkv, qkv, MROWS, NQKV, FEAT);

    // attention + gating -> hcat bf16 [B*S, H*D]
    k_attention<<<BATCH * HEADS, 128, 0, stream>>>(qkv, gate, hcat);

    // proj: [40960,2048] x [2048,512] -> f32
    k_gemm<1><<<dim3(MROWS / BM, DMOD / BN), 256, 0, stream>>>(hcat, wpT, bp, proj, MROWS, DMOD, HD);

    // x1 = LN(proj + x)
    k_ln<true><<<MROWS, 128, 0, stream>>>(proj, x, g1, be1, x1, x1b);

    // ffn1: relu([40960,512] x [512,2048]) -> bf16 (reuses qkv buffer)
    k_gemm<2><<<dim3(MROWS / BM, HD / BN), 256, 0, stream>>>(x1b, w1T, b1, f1, MROWS, HD, DMOD);

    // ffn2: [40960,2048] x [2048,512] -> f32 (reuses proj buffer)
    k_gemm<1><<<dim3(MROWS / BM, DMOD / BN), 256, 0, stream>>>(f1, w2T, b2, ffn, MROWS, DMOD, HD);

    // out = LN(x1 + ffn)
    k_ln<false><<<MROWS, 128, 0, stream>>>(x1, ffn, g2, be2, out, nullptr);
}
